// InvariantPointAttention_7627861918036
// MI455X (gfx1250) — matrix operand order repaired and checked by reading, not compiled, not run on hardware
//
#include <hip/hip_runtime.h>
#include <hip/hip_bf16.h>
#include <math.h>

// ---------------- problem constants ----------------
#define N_RES 768
#define CS    384
#define CZ    128
#define CH    16
#define NH    12        // heads
#define NPQ   4
#define NPV   8
#define INF_  100000.0f
#define EPS_  1e-8f
#define N2    ((size_t)N_RES * N_RES)

typedef __bf16 bf16;
typedef __attribute__((ext_vector_type(16))) __bf16 v16bf;
typedef __attribute__((ext_vector_type(8)))  __bf16 v8bf;
typedef __attribute__((ext_vector_type(8)))  float  v8f;

// ---------------- WMMA helpers (gfx1250, wave32) ----------------
// D(16x16,f32) = A(16x32,bf16) * B(32x16,bf16) + C
__device__ inline v8f wmma_bf16(v16bf a, v16bf b, v8f c) {
  return __builtin_amdgcn_wmma_f32_16x16x32_bf16(false, a, false, b, (short)0, c,
                                                 false, false);
}

// Wide fragment load: lane's 16 bf16 stored contiguously (32 B) -> 2x b128.
__device__ inline v16bf ldv(const bf16* p) { return *(const v16bf*)p; }

// Operand-order index helpers (position of element within a 512-elem tile).
// A tile 16x32 (MxK). ISA: lane<16 -> M=lane, K={0..7,16..23}; lane>=16 -> K={8..15,24..31}
__device__ inline int a_op_idx(int m, int k) {
  int hi = (k >> 3) & 1;
  int j  = ((k >> 4) << 3) | (k & 7);
  return ((hi << 4) + (m & 15)) * 16 + j;
}
// B tile 32x16 (KxN). ISA: lane<16 -> N=lane, K={0..7,16..23}; lane>=16 -> K={8..15,24..31}
__device__ inline int b_op_idx(int k, int n) {
  return ((((k >> 3) & 1) << 4) + (n & 15)) * 16 + (((k >> 4) << 3) | (k & 7));
}
// D/C element r maps to row m = r + 8*(lane>>4), col n = lane&15.

// ---------------- K0: weights prep (hw + operand-swizzled [wb|wdz]) ----------------
__global__ void k0_prep(const float* wb, const float* wdz, const float* head_w,
                        float* hw, bf16* Wop) {
  int t = threadIdx.x;
  if (t < NH) {
    // softplus(hw) * sqrt(1/(3*(PQ*9/2))) = softplus * sqrt(1/54)
    hw[t] = log1pf(__expf(head_w[t])) * sqrtf(1.0f / 54.0f);
  }
  // Wop[kt*3+nt][lane][16] : element j of lane -> (k = kt*32+8*(lane>>4)+(j&7)+16*(j>>3), n = nt*16+(lane&15))
  for (int idx = t; idx < 4 * 3 * 512; idx += blockDim.x) {
    int tile = idx >> 9, rem = idx & 511;
    int kt = tile / 3, nt = tile % 3;
    int lane = rem >> 4, j = rem & 15;
    int k = kt * 32 + ((lane >> 4) << 3) + (j & 7) + ((j >> 3) << 4);
    int n = nt * 16 + (lane & 15);
    float v = 0.f;
    if (n < 12)      v = wb[k * 12 + n];
    else if (n < 44) v = wdz[k * 32 + (n - 12)];
    Wop[idx] = (bf16)v;
  }
}

// ---------------- K1: s-projections + rigid transform + operand packing ----------------
// grid 48, block 256; each block handles 16 residues.
__global__ void k1_proj(const float* s, const float* rot, const float* trans,
                        const float* wq, const float* bq,
                        const float* wkv, const float* bkv,
                        const float* wqp, const float* bqp,
                        const float* wkvp, const float* bkvp,
                        const float* hw,
                        bf16* Qop, bf16* Kop, bf16* Vop,
                        float* cI, float* cJ) {
  __shared__ float s_tile[16 * CS];     // 24 KB
  __shared__ float pbuf[16 * 576];      // 36.9 KB  (point projections)
  int t = threadIdx.x;
  int i0 = blockIdx.x * 16;
  int it = blockIdx.x;
  for (int idx = t; idx < 16 * CS; idx += blockDim.x)
    s_tile[idx] = s[(size_t)(i0 + idx / CS) * CS + (idx % CS)];
  __syncthreads();

  const float S1 = sqrtf(1.0f / (3.0f * CH));

  // Pass A: q (192) + kv (384)
  for (int o = t; o < 576; o += blockDim.x) {
    const float* W; float bv; int col, ld;
    if (o < 192) { W = wq;  bv = bq[o];        col = o;       ld = 192; }
    else         { W = wkv; bv = bkv[o - 192]; col = o - 192; ld = 384; }
    float acc[16];
#pragma unroll
    for (int ii = 0; ii < 16; ++ii) acc[ii] = bv;
    for (int k = 0; k < CS; ++k) {
      float wv = W[(size_t)k * ld + col];
#pragma unroll
      for (int ii = 0; ii < 16; ++ii) acc[ii] += s_tile[ii * CS + k] * wv;
    }
    if (o < 192) {
      int h = o / 16, c = o % 16;
      size_t base = ((size_t)h * 48 + it) * 512;
      for (int ii = 0; ii < 16; ++ii)
        Qop[base + a_op_idx(ii, c)] = (bf16)(acc[ii] * S1);
    } else {
      int oo = o - 192, h = oo / 32, c = oo % 32;
      if (c < 16) {
        size_t base = ((size_t)h * 48 + it) * 512;
        for (int ii = 0; ii < 16; ++ii)
          Kop[base + b_op_idx(c, ii)] = (bf16)acc[ii];
      } else {
        int cc = c - 16;                 // v column 0..15 (ct = 0)
        for (int ii = 0; ii < 16; ++ii) {
          int i = i0 + ii;
          size_t base = (((size_t)h * 24 + (i >> 5)) * 3 + 0) * 512;
          Vop[base + b_op_idx(i & 31, cc)] = (bf16)acc[ii];
        }
      }
    }
  }

  // Pass B: point projections (144 qp + 432 kvp) -> pbuf
  for (int o = t; o < 576; o += blockDim.x) {
    const float* W; float bv; int col, ld;
    if (o < 144) { W = wqp;  bv = bqp[o];        col = o;       ld = 144; }
    else         { W = wkvp; bv = bkvp[o - 144]; col = o - 144; ld = 432; }
    float acc[16];
#pragma unroll
    for (int ii = 0; ii < 16; ++ii) acc[ii] = bv;
    for (int k = 0; k < CS; ++k) {
      float wv = W[(size_t)k * ld + col];
#pragma unroll
      for (int ii = 0; ii < 16; ++ii) acc[ii] += s_tile[ii * CS + k] * wv;
    }
    for (int ii = 0; ii < 16; ++ii) pbuf[ii * 576 + o] = acc[ii];
  }
  __syncthreads();

  // Pass C: rigid transform + augmented dims + separable constants
  if (t < 16 * NH) {
    int ii = t / NH, h = t % NH;
    int i = i0 + ii;
    const float* R = rot + (size_t)i * 9;
    const float* T = trans + (size_t)i * 3;
    float hwh = hw[h];
    size_t qkbase = ((size_t)h * 48 + it) * 512;
    const float* pb = pbuf + ii * 576;
    float qsq = 0.f, ksq = 0.f;
    for (int p = 0; p < NPQ; ++p) {                 // q points
      float lx = pb[0 * 48 + h * 4 + p];
      float ly = pb[1 * 48 + h * 4 + p];
      float lz = pb[2 * 48 + h * 4 + p];
#pragma unroll
      for (int x = 0; x < 3; ++x) {
        float w = T[x] + R[x*3+0]*lx + R[x*3+1]*ly + R[x*3+2]*lz;
        qsq += w * w;
        Qop[qkbase + a_op_idx(ii, 16 + p * 3 + x)] = (bf16)(hwh * w);
      }
    }
    for (int c = 28; c < 32; ++c) Qop[qkbase + a_op_idx(ii, c)] = (bf16)0.0f;
    for (int p = 0; p < NPQ + NPV; ++p) {           // k/v points
      float lx = pb[144 + 0 * 144 + h * 12 + p];
      float ly = pb[144 + 1 * 144 + h * 12 + p];
      float lz = pb[144 + 2 * 144 + h * 12 + p];
      float w3[3];
#pragma unroll
      for (int x = 0; x < 3; ++x)
        w3[x] = T[x] + R[x*3+0]*lx + R[x*3+1]*ly + R[x*3+2]*lz;
      if (p < NPQ) {
#pragma unroll
        for (int x = 0; x < 3; ++x) {
          ksq += w3[x] * w3[x];
          Kop[qkbase + b_op_idx(16 + p * 3 + x, ii)] = (bf16)w3[x];
        }
      } else {
#pragma unroll
        for (int x = 0; x < 3; ++x) {
          int cv = 16 + (p - NPQ) * 3 + x;          // 16..39
          size_t vb = (((size_t)h * 24 + (i >> 5)) * 3 + (cv >> 4)) * 512;
          Vop[vb + b_op_idx(i & 31, cv & 15)] = (bf16)w3[x];
        }
      }
    }
    for (int k = 28; k < 32; ++k) Kop[qkbase + b_op_idx(k, ii)] = (bf16)0.0f;
    for (int cv = 40; cv < 48; ++cv) {
      size_t vb = (((size_t)h * 24 + (i >> 5)) * 3 + (cv >> 4)) * 512;
      Vop[vb + b_op_idx(i & 31, cv & 15)] = (bf16)0.0f;
    }
    cI[i * NH + h] = -0.5f * hwh * qsq;
    cJ[i * NH + h] = -0.5f * hwh * ksq;
  }
}

// ---------------- K2: fused z pass  out[(i,j),0:44] = z(i,j,:) @ [wb|wdz] ----------------
// grid 4608, block 256 (8 waves x 16 rows); streams z (302 MB) exactly once.
// biasT: plain [h][i][j] f32 (contiguous 8-f32 store per lane; coalesced read in K3).
// PzOp: B-operand order [i][kt][ct][lane][16] (lane's 8 D-elems are contiguous -> v8bf store).
__global__ void k2_zproj(const float* z, const float* bb, const float* bdz,
                         const bf16* Wop, float* biasT, bf16* PzOp) {
  __shared__ alignas(32) bf16 Wlds[4 * 3 * 512];    // 12 KB, operand order
  int t = threadIdx.x;
  for (int idx = t; idx < 4 * 3 * 512; idx += blockDim.x) Wlds[idx] = Wop[idx];
  __syncthreads();
  int wave = t >> 5, lane = t & 31;
  size_t r0 = ((size_t)blockIdx.x * 8 + wave) * 16;
  int m = lane & 15;
  const float4* arow4 = (const float4*)(z + (r0 + m) * CZ) + ((lane >> 4) << 1);
  __builtin_prefetch(arow4, 0, 1);                  // global_prefetch_b8
  v8f acc0 = {}, acc1 = {}, acc2 = {};
#pragma unroll
  for (int kt = 0; kt < 4; ++kt) {
    float4 fa = arow4[kt * 8 + 0], fb = arow4[kt * 8 + 1];
    float4 fc = arow4[kt * 8 + 4], fd = arow4[kt * 8 + 5];
    v16bf a;
    a[0]=(bf16)fa.x; a[1]=(bf16)fa.y; a[2]=(bf16)fa.z; a[3]=(bf16)fa.w;
    a[4]=(bf16)fb.x; a[5]=(bf16)fb.y; a[6]=(bf16)fb.z; a[7]=(bf16)fb.w;
    a[8]=(bf16)fc.x; a[9]=(bf16)fc.y; a[10]=(bf16)fc.z; a[11]=(bf16)fc.w;
    a[12]=(bf16)fd.x; a[13]=(bf16)fd.y; a[14]=(bf16)fd.z; a[15]=(bf16)fd.w;
    v16bf b0 = ldv(&Wlds[((kt * 3 + 0) * 32 + lane) * 16]);
    v16bf b1 = ldv(&Wlds[((kt * 3 + 1) * 32 + lane) * 16]);
    v16bf b2 = ldv(&Wlds[((kt * 3 + 2) * 32 + lane) * 16]);
    acc0 = wmma_bf16(a, b0, acc0);
    acc1 = wmma_bf16(a, b1, acc1);
    acc2 = wmma_bf16(a, b2, acc2);
  }
  int n  = lane & 15;
  int mh = (lane >> 4) << 3;
  int i   = (int)(r0 / N_RES);          // constant across the 16-row tile
  int j0t = (int)(r0 % N_RES);          // 16-aligned
  int jb  = j0t + mh;                   // first j of this lane's 8 rows
  if (n < 12) {                         // bias: 8 contiguous f32 per lane
    float bbn = bb[n];
    float4 b0 = make_float4(acc0[0] + bbn, acc0[1] + bbn, acc0[2] + bbn, acc0[3] + bbn);
    float4 b1 = make_float4(acc0[4] + bbn, acc0[5] + bbn, acc0[6] + bbn, acc0[7] + bbn);
    float* bp = biasT + (size_t)n * N2 + (size_t)i * N_RES + jb;
    ((float4*)bp)[0] = b0;
    ((float4*)bp)[1] = b1;
  }
  // pair_z: lane's 8 rows are contiguous bf16 in the B-operand tile
  float bd0 = bdz[n], bd1 = bdz[16 + n];
  v8bf p1, p2;
#pragma unroll
  for (int r = 0; r < 8; ++r) {
    p1[r] = (bf16)(acc1[r] + bd0);
    p2[r] = (bf16)(acc2[r] + bd1);
  }
  int hi = (jb >> 4) & 1;
  size_t pzt = (((size_t)i * 24 + (j0t >> 5)) * 2) * 512;
  size_t pb  = pzt + (size_t)(((((mh >> 3) << 4) + n) << 4) + (hi << 3));
  *(v8bf*)(PzOp + pb)       = p1;
  *(v8bf*)(PzOp + pb + 512) = p2;
}

// ---------------- K3: logits + exp; transpose probs through LDS ----------------
// grid (48, 12), block 32.  Pop[h][it][jt32][lane][16] written as 32-B stores.
__global__ void k3_logits(const bf16* Qop, const bf16* Kop, const float* biasT,
                          const float* cI, const float* cJ, const float* mask,
                          bf16* Pop, float* lsumG) {
  __shared__ float lsum[16];
  __shared__ alignas(32) bf16 ptile[512];           // one A-operand tile staging
  int lane = threadIdx.x & 31;
  int it = blockIdx.x, i0 = it * 16;
  int h  = blockIdx.y;
  if (threadIdx.x < 16) lsum[threadIdx.x] = 0.f;
  __syncthreads();
  v16bf aop = ldv(Qop + ((size_t)h * 48 + it) * 512 + lane * 16);
  int n  = lane & 15;
  int mh = (lane >> 4) << 3;
  float b3 = sqrtf(1.0f / 3.0f);
  float mi[8], cih[8];
#pragma unroll
  for (int r = 0; r < 8; ++r) {
    mi[r]  = mask[i0 + mh + r];
    cih[r] = cI[(i0 + mh + r) * NH + h];
  }
  bf16* popw = Pop + ((size_t)h * 48 + it) * 24 * 512 + (size_t)lane * 16;
  for (int jtp = 0; jtp < 24; ++jtp) {
#pragma unroll
    for (int sub = 0; sub < 2; ++sub) {
      int jt = jtp * 2 + sub;
      v16bf bop = ldv(Kop + ((size_t)h * 48 + jt) * 512 + lane * 16);
      v8f d = {};
      d = wmma_bf16(aop, bop, d);
      int j = jt * 16 + n;
      float cj = cJ[j * NH + h];
      float mj = mask[j];
      const float* brow = biasT + (size_t)h * N2 + j;
      int kk    = (sub << 4) + n;                   // j within its 32-tile
      int hi    = (kk >> 3) & 1;
      int jelem = ((kk >> 4) << 3) | (kk & 7);
      bf16* lw = ptile + (hi << 8) + jelem;
#pragma unroll
      for (int r = 0; r < 8; ++r) {
        int i = i0 + mh + r;
        float logit = d[r] + b3 * brow[(size_t)i * N_RES] + cih[r] + cj
                    + INF_ * (mi[r] * mj - 1.0f);
        float e = __expf(logit);
        atomicAdd(&lsum[mh + r], e);
        lw[(mh + r) * 16] = (bf16)e;
      }
    }
    __syncthreads();
    *(v16bf*)(popw + (size_t)jtp * 512) = ldv(&ptile[lane * 16]);  // 32-B store
    __syncthreads();
  }
  if (threadIdx.x < 16)
    lsumG[h * N_RES + i0 + threadIdx.x] = lsum[threadIdx.x];
}

// ---------------- K4: o / o_pt gemm:  probs(16x768) @ Vhat(768x48) ----------------
// grid (48, 3, 12), block 32.  All fragment loads contiguous 32 B per lane.
__global__ void k4_av(const bf16* Pop, const bf16* Vop, const float* lsumG,
                      float* o_raw) {
  int lane = threadIdx.x & 31;
  int i0 = blockIdx.x * 16, ct = blockIdx.y, h = blockIdx.z;
  const bf16* abase = Pop + ((size_t)h * 48 + blockIdx.x) * 24 * 512 + (size_t)lane * 16;
  const bf16* bbase = Vop + ((size_t)h * 24 * 3 + ct) * 512 + (size_t)lane * 16;
  v8f acc = {};
  for (int kt = 0; kt < 24; ++kt) {
    __builtin_prefetch(abase + (size_t)(kt + 1) * 512, 0, 1);
    v16bf a = ldv(abase + (size_t)kt * 512);
    v16bf b = ldv(bbase + (size_t)kt * 3 * 512);
    acc = wmma_bf16(a, b, acc);
  }
  int n = lane & 15, mh = (lane >> 4) << 3;
#pragma unroll
  for (int r = 0; r < 8; ++r) {
    int i = i0 + mh + r;
    float l = lsumG[h * N_RES + i];
    o_raw[(size_t)i * (NH * 48) + h * 48 + ct * 16 + n] = acc[r] / l;
  }
}

// ---------------- K5: o_pair gemm:  probs(12hx768) @ pair_z[i](768x32) ----------------
// grid (768, 2), block 32.  Head-row A-fragments are contiguous in Pop layout.
__global__ void k5_opair(const bf16* Pop, const bf16* PzOp, const float* lsumG,
                         float* opair) {
  int lane = threadIdx.x & 31;
  int i = blockIdx.x, ct = blockIdx.y;
  int m  = lane & 15;           // head row
  int hl = lane >> 4;
  const bf16* apb = Pop + ((size_t)m * 48 + (i >> 4)) * 24 * 512
                  + (size_t)(((hl << 4) + (i & 15)) * 16);
  const bf16* bpb = PzOp + ((size_t)i * 24 * 2 + ct) * 512 + (size_t)lane * 16;
  v8f acc = {};
  for (int kt = 0; kt < 24; ++kt) {
    v16bf a;
    if (m < NH) a = ldv(apb + (size_t)kt * 512);
    else        { v16bf zv = {}; a = zv; }
    v16bf b = ldv(bpb + (size_t)kt * 2 * 512);
    acc = wmma_bf16(a, b, acc);
  }
  int n = lane & 15, mh = (lane >> 4) << 3;
#pragma unroll
  for (int r = 0; r < 8; ++r) {
    int hh = mh + r;
    if (hh < NH) {
      float l = lsumG[hh * N_RES + i];
      opair[(size_t)i * (NH * 32) + hh * 32 + ct * 16 + n] = acc[r] / l;
    }
  }
}

// ---------------- K6: inverse transform, norms, concat, final projection ----------------
// grid 48, block 256 (16 residues per block so wout streams ~once per tile).
__global__ void k6_final(const float* o_raw, const float* opair,
                         const float* rot, const float* trans,
                         const float* wout, const float* bout, float* out) {
  __shared__ float f[16 * 960];         // 61.4 KB
  int t = threadIdx.x;
  int i0 = blockIdx.x * 16;
  for (int idx = t; idx < 16 * 960; idx += blockDim.x) {
    int ii = idx / 960, c = idx % 960;
    int i = i0 + ii;
    float v;
    if (c < 192) {
      v = o_raw[(size_t)i * 576 + (c >> 4) * 48 + (c & 15)];
    } else if (c < 576) {
      int cc = c - 192, comp = cc / 96, pt = cc % 96;
      int h = pt >> 3, p = pt & 7;
      const float* pr = o_raw + (size_t)i * 576 + h * 48 + 16 + p * 3;
      const float* R = rot + (size_t)i * 9;
      const float* T = trans + (size_t)i * 3;
      float gx = pr[0] - T[0], gy = pr[1] - T[1], gz = pr[2] - T[2];
      float lx = R[0]*gx + R[3]*gy + R[6]*gz;   // R^T * g
      float ly = R[1]*gx + R[4]*gy + R[7]*gz;
      float lz = R[2]*gx + R[5]*gy + R[8]*gz;
      v = (comp == 0) ? lx : (comp == 1) ? ly : (comp == 2) ? lz
        : sqrtf(lx*lx + ly*ly + lz*lz + EPS_);
    } else {
      v = opair[(size_t)i * 384 + (c - 576)];
    }
    f[idx] = v;
  }
  __syncthreads();
  for (int idx = t; idx < 16 * CS; idx += blockDim.x) {
    int ii = idx / CS, o = idx % CS;
    float acc = bout[o];
    const float* fr = &f[ii * 960];
    for (int k = 0; k < 960; ++k) acc += fr[k] * wout[(size_t)k * CS + o];
    out[(size_t)(i0 + ii) * CS + o] = acc;
  }
}

// ---------------- workspace layout ----------------
static constexpr size_t aln(size_t x) { return (x + 255) & ~(size_t)255; }
static constexpr size_t OFF_HW    = 0;
static constexpr size_t OFF_WOP   = aln(OFF_HW    + 16 * 4);
static constexpr size_t OFF_QOP   = aln(OFF_WOP   + (size_t)4 * 3 * 512 * 2);
static constexpr size_t OFF_KOP   = aln(OFF_QOP   + (size_t)NH * 48 * 512 * 2);
static constexpr size_t OFF_VOP   = aln(OFF_KOP   + (size_t)NH * 48 * 512 * 2);
static constexpr size_t OFF_CI    = aln(OFF_VOP   + (size_t)NH * 24 * 3 * 512 * 2);
static constexpr size_t OFF_CJ    = aln(OFF_CI    + (size_t)N_RES * NH * 4);
static constexpr size_t OFF_BIAS  = aln(OFF_CJ    + (size_t)N_RES * NH * 4);
static constexpr size_t OFF_PZOP  = aln(OFF_BIAS  + (size_t)NH * N_RES * N_RES * 4);
static constexpr size_t OFF_POP   = aln(OFF_PZOP  + (size_t)N_RES * 24 * 2 * 512 * 2);
static constexpr size_t OFF_LSUM  = aln(OFF_POP   + (size_t)NH * 48 * 24 * 512 * 2);
static constexpr size_t OFF_ORAW  = aln(OFF_LSUM  + (size_t)NH * N_RES * 4);
static constexpr size_t OFF_OPAIR = aln(OFF_ORAW  + (size_t)N_RES * NH * 48 * 4);
// total ~85 MB

extern "C" void kernel_launch(void* const* d_in, const int* in_sizes, int n_in,
                              void* d_out, int out_size, void* d_ws, size_t ws_size,
                              hipStream_t stream) {
  (void)in_sizes; (void)n_in; (void)out_size; (void)ws_size;
  const float* s     = (const float*)d_in[0];
  const float* z     = (const float*)d_in[1];
  const float* rot   = (const float*)d_in[2];
  const float* trans = (const float*)d_in[3];
  const float* mask  = (const float*)d_in[4];
  const float* wq    = (const float*)d_in[5];
  const float* bq    = (const float*)d_in[6];
  const float* wkv   = (const float*)d_in[7];
  const float* bkv   = (const float*)d_in[8];
  const float* wqp   = (const float*)d_in[9];
  const float* bqp   = (const float*)d_in[10];
  const float* wkvp  = (const float*)d_in[11];
  const float* bkvp  = (const float*)d_in[12];
  const float* wb    = (const float*)d_in[13];
  const float* bb    = (const float*)d_in[14];
  const float* wdz   = (const float*)d_in[15];
  const float* bdz   = (const float*)d_in[16];
  const float* hwp   = (const float*)d_in[17];
  const float* wout  = (const float*)d_in[18];
  const float* bout  = (const float*)d_in[19];
  char* w = (char*)d_ws;

  k0_prep<<<1, 256, 0, stream>>>(wb, wdz, hwp,
      (float*)(w + OFF_HW), (bf16*)(w + OFF_WOP));

  k1_proj<<<48, 256, 0, stream>>>(s, rot, trans, wq, bq, wkv, bkv, wqp, bqp,
      wkvp, bkvp, (const float*)(w + OFF_HW),
      (bf16*)(w + OFF_QOP), (bf16*)(w + OFF_KOP), (bf16*)(w + OFF_VOP),
      (float*)(w + OFF_CI), (float*)(w + OFF_CJ));

  k2_zproj<<<4608, 256, 0, stream>>>(z, bb, bdz, (const bf16*)(w + OFF_WOP),
      (float*)(w + OFF_BIAS), (bf16*)(w + OFF_PZOP));

  k3_logits<<<dim3(48, 12), 32, 0, stream>>>(
      (const bf16*)(w + OFF_QOP), (const bf16*)(w + OFF_KOP),
      (const float*)(w + OFF_BIAS), (const float*)(w + OFF_CI),
      (const float*)(w + OFF_CJ), mask,
      (bf16*)(w + OFF_POP), (float*)(w + OFF_LSUM));

  k4_av<<<dim3(48, 3, 12), 32, 0, stream>>>(
      (const bf16*)(w + OFF_POP), (const bf16*)(w + OFF_VOP),
      (const float*)(w + OFF_LSUM), (float*)(w + OFF_ORAW));

  k5_opair<<<dim3(768, 2), 32, 0, stream>>>(
      (const bf16*)(w + OFF_POP), (const bf16*)(w + OFF_PZOP),
      (const float*)(w + OFF_LSUM), (float*)(w + OFF_OPAIR));

  k6_final<<<48, 256, 0, stream>>>(
      (const float*)(w + OFF_ORAW), (const float*)(w + OFF_OPAIR),
      rot, trans, wout, bout, (float*)d_out);
}
